// CacheAugmentation_65747359367688
// MI455X (gfx1250) — compile-verified, matrix operand order checked by reading, not run on hardware
//
#include <hip/hip_runtime.h>
#include <hip/hip_bf16.h>

typedef __attribute__((ext_vector_type(16))) __bf16 v16bf;
typedef __attribute__((ext_vector_type(8)))  __bf16 v8bf;
typedef __attribute__((ext_vector_type(8)))  float  v8f;

union FragU { v16bf v; v8bf h[2]; };

#define HD     96
#define NCACHE 2048
#define HDIM   768
#define SLEN   4096
#define BATCH  2

#define LDK 104   // K-tile LDS leading dim (elements); 208B rows, 16B aligned
#define LDV 40    // V-tile LDS leading dim (elements); 80B rows, 16B aligned

// ---------------------------------------------------------------------------
// Fragment load: row-major slab starting at `base` (element [row0,col0]),
// leading dimension ld. Produces the CDNA5 16-bit A-fragment (16 rows x 32 K);
// identical per-lane pattern serves as the B-fragment of the [NxK] row-major
// transpose.
//   lane l<16 : row=l,    k = {0..7, 16..23}
//   lane l>=16: row=l-16, k = {8..15, 24..31}
// ---------------------------------------------------------------------------
__device__ __forceinline__ v16bf load_frag(const __bf16* base, int ld, int lane) {
  const int lr = lane & 15, hf = lane >> 4;
  const __bf16* p = base + (size_t)lr * ld + hf * 8;
  FragU f;
  f.h[0] = *(const v8bf*)(p);
  f.h[1] = *(const v8bf*)(p + 16);
  return f.v;
}

__device__ __forceinline__ v8f wmma_bf16(v16bf a, v16bf b, v8f c) {
  return __builtin_amdgcn_wmma_f32_16x16x32_bf16(false, a, false, b, (short)0, c,
                                                 false, false);
}

// reductions across the 16 lanes of a wave32 half (rows of a C-fragment)
__device__ __forceinline__ float redmax16(float v) {
#pragma unroll
  for (int o = 8; o; o >>= 1) v = fmaxf(v, __shfl_xor(v, o, 32));
  return v;
}
__device__ __forceinline__ float redsum16(float v) {
#pragma unroll
  for (int o = 8; o; o >>= 1) v += __shfl_xor(v, o, 32);
  return v;
}

// ---------------------------------------------------------------------------
// elementwise f32 -> bf16 cast
// ---------------------------------------------------------------------------
__global__ void k_cast(const float* __restrict__ in, __bf16* __restrict__ out, int n) {
  int i = blockIdx.x * blockDim.x + threadIdx.x;
  int stride = gridDim.x * blockDim.x;
  for (; i < n; i += stride) out[i] = (__bf16)in[i];
}

// cvT[h][d][n] = cache_values[n][h*HD + d]   (bf16)
__global__ void k_cvT(const float* __restrict__ cv, __bf16* __restrict__ o) {
  int i = blockIdx.x * blockDim.x + threadIdx.x;
  if (i >= 8 * HD * NCACHE) return;
  int n = i & (NCACHE - 1);
  int rest = i >> 11;
  int d = rest % HD;
  int h = rest / HD;
  o[i] = (__bf16)cv[(size_t)n * HDIM + h * HD + d];
}

// ---------------------------------------------------------------------------
// C = A * B^T + bias  (bf16 out).  A:[M,K] bf16, B:[N,K] bf16 row-major.
// One wave per block; 32(M) x 64(N) tile: 2 A-frags x 4 B-frags = 8 WMMAs
// per 12 b128 loads per K-step.
// ---------------------------------------------------------------------------
__global__ void k_gemm_q(const __bf16* __restrict__ A, const __bf16* __restrict__ B,
                         const float* __restrict__ bias, __bf16* __restrict__ C,
                         int M, int N, int K) {
  const int lane = threadIdx.x & 31;
  const int m0 = blockIdx.x * 32;
  const int n0 = blockIdx.y * 64;
  v8f acc[2][4] = {};
  for (int kb = 0; kb < K; kb += 32) {
    v16bf a0 = load_frag(A + (size_t)m0 * K + kb, K, lane);
    v16bf a1 = load_frag(A + (size_t)(m0 + 16) * K + kb, K, lane);
#pragma unroll
    for (int j = 0; j < 4; ++j) {
      v16bf b = load_frag(B + (size_t)(n0 + 16 * j) * K + kb, K, lane);
      acc[0][j] = wmma_bf16(a0, b, acc[0][j]);
      acc[1][j] = wmma_bf16(a1, b, acc[1][j]);
    }
  }
  const int lr = lane & 15, hf = lane >> 4;
#pragma unroll
  for (int i = 0; i < 2; ++i)
#pragma unroll
    for (int j = 0; j < 4; ++j) {
      int col = n0 + 16 * j + lr;
      float bv = bias[col];
#pragma unroll
      for (int r = 0; r < 8; ++r) {
        int row = m0 + 16 * i + hf * 8 + r;
        C[(size_t)row * N + col] = (__bf16)(acc[i][j][r] + bv);
      }
    }
}

// y = res + A * B^T + bias   (f32 out) — output projection + residual
__global__ void k_gemm_o(const __bf16* __restrict__ A, const __bf16* __restrict__ B,
                         const float* __restrict__ bias, const float* __restrict__ res,
                         float* __restrict__ C, int M, int N, int K) {
  const int lane = threadIdx.x & 31;
  const int m0 = blockIdx.x * 32;
  const int n0 = blockIdx.y * 64;
  v8f acc[2][4] = {};
  for (int kb = 0; kb < K; kb += 32) {
    v16bf a0 = load_frag(A + (size_t)m0 * K + kb, K, lane);
    v16bf a1 = load_frag(A + (size_t)(m0 + 16) * K + kb, K, lane);
#pragma unroll
    for (int j = 0; j < 4; ++j) {
      v16bf b = load_frag(B + (size_t)(n0 + 16 * j) * K + kb, K, lane);
      acc[0][j] = wmma_bf16(a0, b, acc[0][j]);
      acc[1][j] = wmma_bf16(a1, b, acc[1][j]);
    }
  }
  const int lr = lane & 15, hf = lane >> 4;
#pragma unroll
  for (int i = 0; i < 2; ++i)
#pragma unroll
    for (int j = 0; j < 4; ++j) {
      int col = n0 + 16 * j + lr;
      float bv = bias[col];
#pragma unroll
      for (int r = 0; r < 8; ++r) {
        size_t idx = (size_t)(m0 + 16 * i + hf * 8 + r) * N + col;
        C[idx] = res[idx] + bv + acc[i][j][r];
      }
    }
}

// ---------------------------------------------------------------------------
// Flash attention over the cache bank.
// Block = 8 waves; all 8 waves share one (b,h) (8 divides the 256 S-tiles per
// head), so K/V chunks are staged into LDS once per block (double-buffered)
// and every wave reads its WMMA B-fragments from LDS (8x less L2 traffic).
// Each wave owns one 16-row S-tile; N streamed in chunks of 32.
// ---------------------------------------------------------------------------
__global__ void k_attn(const __bf16* __restrict__ q, const __bf16* __restrict__ ck,
                       const __bf16* __restrict__ cvT, const float* __restrict__ age,
                       __bf16* __restrict__ ctx) {
  __shared__ __bf16 lds_k[2][32 * LDK];  // 32 rows x 96 K, double buffered
  __shared__ __bf16 lds_v[2][HD * LDV];  // 96 rows x 32 N, double buffered
  __shared__ __bf16 lds_p[8][16 * 48];   // per-wave 16x32 P tile

  const int tid = threadIdx.x;
  const int lane = tid & 31;
  const int wv = tid >> 5;
  const int tile = blockIdx.x * 8 + wv;      // B*NH*(S/16) = 4096 tiles
  const int b = tile >> 11;
  const int h = (tile >> 8) & 7;             // same for all 8 waves in a block
  const int s0 = (tile & 255) * 16;
  const int lr = lane & 15, hf = lane >> 4;
  const float scale = 0.10206207261596575f;  // 1/sqrt(96)

  const __bf16* kbase = ck + h * HD;
  const __bf16* vbase = cvT + (size_t)h * HD * NCACHE;

  // cooperative stage of one K-chunk (32x96) and V-chunk (96x32) into LDS
  auto stage = [&](int bb, int n0s) {
#pragma unroll
    for (int t = tid; t < 384; t += 256) {   // K: 32 rows x 12 x v8bf
      int row = t / 12, c = (t % 12) * 8;
      *(v8bf*)(&lds_k[bb][row * LDK + c]) =
          *(const v8bf*)(kbase + (size_t)(n0s + row) * HDIM + c);
    }
#pragma unroll
    for (int t = tid; t < 384; t += 256) {   // V: 96 rows x 4 x v8bf
      int row = t >> 2, c = (t & 3) * 8;
      *(v8bf*)(&lds_v[bb][row * LDV + c]) =
          *(const v8bf*)(vbase + (size_t)row * NCACHE + n0s + c);
    }
  };

  v8f acc[6] = {};
  float m[8], l[8];
#pragma unroll
  for (int r = 0; r < 8; ++r) { m[r] = -3.0e38f; l[r] = 0.f; }

  // Q tile is loop-invariant: hoist all 3 fragments (16 x 96) into registers.
  const __bf16* qbase = q + ((size_t)(b * SLEN + s0)) * HDIM + h * HD;
  v16bf qf[3];
#pragma unroll
  for (int kk = 0; kk < 3; ++kk) qf[kk] = load_frag(qbase + kk * 32, HDIM, lane);

  stage(0, 0);
  int buf = 0;

  for (int n0 = 0; n0 < NCACHE; n0 += 32) {
    __syncthreads();  // stage(buf) visible; previous reads of buf^1 retired
    if (n0 + 32 < NCACHE) stage(buf ^ 1, n0 + 32);
    // prefetch the chunk after next into L2 (global_prefetch_b8)
    if (n0 + 64 < NCACHE) {
      __builtin_prefetch(kbase + (size_t)(n0 + 64 + lane) * HDIM, 0, 0);
      __builtin_prefetch(vbase + (size_t)(lane * 3) * NCACHE + n0 + 64, 0, 0);
    }

    // ---- scores: 16(S) x 32(N) tile, K = hd = 96, B-frags from LDS ----
    v8f sc[2] = {};
#pragma unroll
    for (int kk = 0; kk < 3; ++kk)
#pragma unroll
      for (int j = 0; j < 2; ++j) {
        v16bf bb = load_frag(&lds_k[buf][(16 * j) * LDK + kk * 32], LDK, lane);
        sc[j] = wmma_bf16(qf[kk], bb, sc[j]);
      }

    float pen0 = 0.1f * age[n0 + lr];
    float pen1 = 0.1f * age[n0 + 16 + lr];

    // ---- online softmax (per-row stats across the 16 lanes of each half) ----
    float p0[8], p1[8], alpha[8];
#pragma unroll
    for (int r = 0; r < 8; ++r) {
      float v0 = sc[0][r] * scale - pen0;
      float v1 = sc[1][r] * scale - pen1;
      float t = redmax16(fmaxf(v0, v1));
      float mn = fmaxf(m[r], t);
      alpha[r] = __expf(m[r] - mn);
      m[r] = mn;
      p0[r] = __expf(v0 - mn);
      p1[r] = __expf(v1 - mn);
      l[r] = l[r] * alpha[r] + redsum16(p0[r] + p1[r]);
    }
#pragma unroll
    for (int t = 0; t < 6; ++t)
#pragma unroll
      for (int r = 0; r < 8; ++r) acc[t][r] *= alpha[r];

    // ---- C-layout P -> LDS -> A-layout fragment (private per wave) ----
#pragma unroll
    for (int r = 0; r < 8; ++r) {
      int row = hf * 8 + r;
      lds_p[wv][row * 48 + lr]      = (__bf16)p0[r];
      lds_p[wv][row * 48 + 16 + lr] = (__bf16)p1[r];
    }
    asm volatile("s_wait_dscnt 0x0" ::: "memory");
    FragU pf;
    {
      const __bf16* lp = &lds_p[wv][(size_t)lr * 48 + hf * 8];
      pf.h[0] = *(const v8bf*)(lp);
      pf.h[1] = *(const v8bf*)(lp + 16);
    }
    asm volatile("" ::: "memory");

    // ---- acc(16x96) += P(16x32) * V(32x96), B-frags from LDS ----
#pragma unroll
    for (int t = 0; t < 6; ++t) {
      v16bf bb = load_frag(&lds_v[buf][(t * 16) * LDV], LDV, lane);
      acc[t] = wmma_bf16(pf.v, bb, acc[t]);
    }
    buf ^= 1;
  }

#pragma unroll
  for (int r = 0; r < 8; ++r) l[r] = 1.0f / l[r];
#pragma unroll
  for (int t = 0; t < 6; ++t) {
#pragma unroll
    for (int r = 0; r < 8; ++r) {
      int row = s0 + hf * 8 + r;
      ctx[((size_t)(b * SLEN + row)) * HDIM + h * HD + t * 16 + lr] =
          (__bf16)(acc[t][r] * l[r]);
    }
  }
}

// ---------------------------------------------------------------------------
// LayerNorm over H=768; one block per row, 3 elements per thread.
// ---------------------------------------------------------------------------
__global__ void k_ln(const float* __restrict__ y, const float* __restrict__ g,
                     const float* __restrict__ bt, float* __restrict__ out) {
  __shared__ float red[256];
  const int row = blockIdx.x;
  const int t = threadIdx.x;
  const float* yr = y + (size_t)row * HDIM;
  float x0 = yr[t], x1 = yr[t + 256], x2 = yr[t + 512];
  red[t] = x0 + x1 + x2;
  __syncthreads();
  for (int o = 128; o; o >>= 1) { if (t < o) red[t] += red[t + o]; __syncthreads(); }
  float mu = red[0] * (1.0f / HDIM);
  __syncthreads();
  float d0 = x0 - mu, d1 = x1 - mu, d2 = x2 - mu;
  red[t] = d0 * d0 + d1 * d1 + d2 * d2;
  __syncthreads();
  for (int o = 128; o; o >>= 1) { if (t < o) red[t] += red[t + o]; __syncthreads(); }
  float rstd = rsqrtf(red[0] * (1.0f / HDIM) + 1e-5f);
  float* orow = out + (size_t)row * HDIM;
  orow[t]       = d0 * rstd * g[t]       + bt[t];
  orow[t + 256] = d1 * rstd * g[t + 256] + bt[t + 256];
  orow[t + 512] = d2 * rstd * g[t + 512] + bt[t + 512];
}

// ---------------------------------------------------------------------------
extern "C" void kernel_launch(void* const* d_in, const int* in_sizes, int n_in,
                              void* d_out, int out_size, void* d_ws, size_t ws_size,
                              hipStream_t stream) {
  (void)in_sizes; (void)n_in; (void)out_size; (void)ws_size;
  const float* inputs       = (const float*)d_in[0];
  const float* Wq           = (const float*)d_in[1];
  const float* bq           = (const float*)d_in[2];
  const float* Wo           = (const float*)d_in[3];
  const float* bo           = (const float*)d_in[4];
  const float* cache_keys   = (const float*)d_in[5];
  const float* cache_values = (const float*)d_in[6];
  const float* cache_age    = (const float*)d_in[7];
  const float* ln_g         = (const float*)d_in[8];
  const float* ln_b         = (const float*)d_in[9];
  float* out = (float*)d_out;

  const int M = BATCH * SLEN;  // 8192
  char* w = (char*)d_ws;
  size_t off = 0;
  auto alloc = [&](size_t bytes) -> char* {
    char* p = w + off;
    off += (bytes + 255) & ~(size_t)255;
    return p;
  };
  __bf16* Xbf  = (__bf16*)alloc((size_t)M * HDIM * 2);
  __bf16* Wqb  = (__bf16*)alloc((size_t)HDIM * HDIM * 2);
  __bf16* Wob  = (__bf16*)alloc((size_t)HDIM * HDIM * 2);
  __bf16* ckb  = (__bf16*)alloc((size_t)NCACHE * HDIM * 2);
  __bf16* cvT  = (__bf16*)alloc((size_t)8 * HD * NCACHE * 2);
  __bf16* qb   = (__bf16*)alloc((size_t)M * HDIM * 2);
  __bf16* ctxb = (__bf16*)alloc((size_t)M * HDIM * 2);
  float*  ybuf = (float*)alloc((size_t)M * HDIM * 4);

  k_cast<<<1024, 256, 0, stream>>>(inputs, Xbf, M * HDIM);
  k_cast<<<256, 256, 0, stream>>>(Wq, Wqb, HDIM * HDIM);
  k_cast<<<256, 256, 0, stream>>>(Wo, Wob, HDIM * HDIM);
  k_cast<<<512, 256, 0, stream>>>(cache_keys, ckb, NCACHE * HDIM);
  k_cvT<<<6144, 256, 0, stream>>>(cache_values, cvT);

  k_gemm_q<<<dim3(M / 32, HDIM / 64), 32, 0, stream>>>(Xbf, Wqb, bq, qb, M, HDIM, HDIM);
  k_attn<<<512, 256, 0, stream>>>(qb, ckb, cvT, cache_age, ctxb);
  k_gemm_o<<<dim3(M / 32, HDIM / 64), 32, 0, stream>>>(ctxb, Wob, bo, inputs, ybuf,
                                                       M, HDIM, HDIM);
  k_ln<<<M, 256, 0, stream>>>(ybuf, ln_g, ln_b, out);
}
